// MS2FN_18640158064957
// MI455X (gfx1250) — matrix operand order repaired and checked by reading, not compiled, not run on hardware
//
#include <hip/hip_runtime.h>
#include <cstdint>

typedef _Float16 half_t;
typedef __attribute__((ext_vector_type(16))) _Float16     v16h;
typedef __attribute__((ext_vector_type(8)))  float        v8f;
typedef __attribute__((ext_vector_type(8)))  unsigned int v8u;

__device__ __forceinline__ float mishf(float v) {
  float sp = (v > 20.f) ? v : log1pf(expf(v));
  return v * tanhf(sp);
}

// ------------------------------------------------------------------
// Generic WMMA GEMM:  C[M,N](f32) = A[M,K](f16) * B[K,N](f16) (+bias[n])
// Batched via blockIdx.z with element strides sA/sB/sC.
// convMode=1: A is an implicit im2col of a (imgs, cL, cL, 128) channels-last
// tensor; k -> (tap, c=k&127), tap -> (dy,dx); row -> (img,y,x); zero pad.
//
// 256 threads = 8 waves; 64x64 C tile; each wave: 16x32 (two 16x16x32 wmma).
// LDS is double-buffered. A rows are stored in plain k order (identity dword
// order) and B is stored transposed (column-major), so each lane's
// 16 fragment halves are one contiguous, 16B-aligned 32-byte run ->
// the compiler emits ds_load_b128 pairs instead of 16 ds_load_u16.
// ------------------------------------------------------------------
#define AST 40   // halves per A row  (32 used + pad; 80B keeps 16B alignment)
#define BST 40   // halves per B column

__global__ void ms2fn_gemm_wmma(
    const half_t* __restrict__ A, const half_t* __restrict__ Bm,
    const float* __restrict__ bias, float* __restrict__ C, half_t* __restrict__ Ch,
    int M, int N, int K, int lda, int ldb, int ldc,
    long long sA, long long sB, long long sC,
    int convMode, int cL, int cK, int cPad)
{
  __shared__ half_t As[2][64 * AST];
  __shared__ half_t Bs[2][64 * BST];

  const half_t* Ap = A + (long long)blockIdx.z * sA;
  const half_t* Bp = Bm + (long long)blockIdx.z * sB;

  const int tid  = threadIdx.x;
  const int lane = tid & 31;
  const int wave = tid >> 5;      // 0..7
  const int wm   = wave & 3;      // which 16-row tile
  const int wn   = wave >> 2;     // which 32-col tile
  const int hlf  = lane >> 4;     // lane half (0/1)
  const int l16  = lane & 15;

  const int m0 = blockIdx.y * 64;
  const int n0 = blockIdx.x * 64;

  // stage one 64x32 A tile + 32x64 B tile into LDS buffer `buf`
  auto stage = [&](int k0, int buf) {
    for (int idx = tid; idx < 64 * 32; idx += 256) {
      int mi = idx >> 5, ki = idx & 31;
      int r = m0 + mi, kk = k0 + ki;
      half_t v = (half_t)0.f;
      if (r < M && kk < K) {
        if (convMode) {
          unsigned tap = (unsigned)kk >> 7;      // cC == 128 always
          int c  = kk & 127;
          unsigned dy = tap / (unsigned)cK, dx = tap - dy * (unsigned)cK;
          unsigned ll = (unsigned)(cL * cL);
          unsigned img = (unsigned)r / ll;
          unsigned pix = (unsigned)r - img * ll;
          int iy = (int)(pix / (unsigned)cL) + (int)dy - cPad;
          int ix = (int)(pix % (unsigned)cL) + (int)dx - cPad;
          if (iy >= 0 && iy < cL && ix >= 0 && ix < cL)
            v = Ap[(((long long)img * cL + iy) * cL + ix) * 128 + c];
        } else {
          v = Ap[(long long)r * lda + kk];
        }
      }
      // identity dword order (contiguous k, matching transposed B):
      // lane half 0 needs K-dwords {0..7} at positions 0..7
      // lane half 1 needs K-dwords {8..15} at positions 8..15
      int d = ki >> 1;
      int pos = d;
      As[buf][mi * AST + pos * 2 + (ki & 1)] = v;
    }
    for (int idx = tid; idx < 32 * 64; idx += 256) {
      int ki = idx >> 6, ni = idx & 63;     // ni fastest -> coalesced global read
      int kk = k0 + ki, n = n0 + ni;
      half_t v = (kk < K && n < N) ? Bp[(long long)kk * ldb + n] : (half_t)0.f;
      Bs[buf][ni * BST + ki] = v;           // transposed: K contiguous per column
    }
  };

  v8f acc0 = {}; v8f acc1 = {};
  int buf = 0;
  stage(0, 0);
  __syncthreads();

  for (int k0 = 0; k0 < K; k0 += 32) {
    if (k0 + 32 < K) stage(k0 + 32, buf ^ 1);

    v16h a  = __builtin_bit_cast(v16h,
        *(const v8u*)(&As[buf][(wm * 16 + l16) * AST + hlf * 16]));
    v16h b0 = __builtin_bit_cast(v16h,
        *(const v8u*)(&Bs[buf][(wn * 32 + l16) * BST + hlf * 16]));
    v16h b1 = __builtin_bit_cast(v16h,
        *(const v8u*)(&Bs[buf][(wn * 32 + 16 + l16) * BST + hlf * 16]));

    acc0 = __builtin_amdgcn_wmma_f32_16x16x32_f16(false, a, false, b0, (short)0, acc0, false, false);
    acc1 = __builtin_amdgcn_wmma_f32_16x16x32_f16(false, a, false, b1, (short)0, acc1, false, false);

    __syncthreads();
    buf ^= 1;
  }

  // C layout: VGPR v -> M = v + 8*half, N = lane%16
  float*  Cp  = C + (long long)blockIdx.z * sC;
  half_t* Chp = Ch ? (Ch + (long long)blockIdx.z * sC) : nullptr;
  #pragma unroll
  for (int v = 0; v < 8; ++v) {
    int m = m0 + wm * 16 + v + hlf * 8;
    if (m >= M) continue;
    int na = n0 + wn * 32 + l16;
    if (na < N) {
      float val = acc0[v] + (bias ? bias[na] : 0.f);
      Cp[(long long)m * ldc + na] = val;
      if (Chp) Chp[(long long)m * ldc + na] = (half_t)val;
    }
    int nb = na + 16;
    if (nb < N) {
      float val = acc1[v] + (bias ? bias[nb] : 0.f);
      Cp[(long long)m * ldc + nb] = val;
      if (Chp) Chp[(long long)m * ldc + nb] = (half_t)val;
    }
  }
}

// ---- BN statistics: channel c owns elements i = ((a*mod + c) << dsh) + d ----
// one block per channel; div == 1<<dsh (dsh in {0,7}); all counts < 2^31.
__global__ void ms2fn_stats(const float* __restrict__ x, float* __restrict__ mean,
                            float* __restrict__ var, unsigned total, int dsh, int mod)
{
  int c = blockIdx.x;
  unsigned per = total / (unsigned)mod;
  unsigned dmask = (1u << dsh) - 1u;
  float s = 0.f, s2 = 0.f;
  for (unsigned t = threadIdx.x; t < per; t += blockDim.x) {
    unsigned a = t >> dsh, d = t & dmask;
    unsigned i = (((a * (unsigned)mod + (unsigned)c)) << dsh) + d;
    float v = x[i]; s += v; s2 += v * v;
  }
  __shared__ float sh[256], sh2[256];
  sh[threadIdx.x] = s; sh2[threadIdx.x] = s2; __syncthreads();
  for (int o = 128; o > 0; o >>= 1) {
    if (threadIdx.x < o) { sh[threadIdx.x] += sh[threadIdx.x + o]; sh2[threadIdx.x] += sh2[threadIdx.x + o]; }
    __syncthreads();
  }
  if (threadIdx.x == 0) {
    float m = sh[0] / (float)per;
    mean[c] = m; var[c] = sh2[0] / (float)per - m * m;
  }
}

// grid = (cdiv(per,256), mod); no runtime division anywhere.
__global__ void ms2fn_bn_mish(const float* __restrict__ x, float* __restrict__ y,
                              half_t* __restrict__ yh,
                              const float* __restrict__ mean, const float* __restrict__ var,
                              const float* __restrict__ g, const float* __restrict__ be,
                              unsigned per, int dsh, int mod)
{
  unsigned t = blockIdx.x * blockDim.x + threadIdx.x;
  if (t >= per) return;
  int c = blockIdx.y;
  unsigned a = t >> dsh, d = t & ((1u << dsh) - 1u);
  unsigned i = ((a * (unsigned)mod + (unsigned)c) << dsh) + d;
  float v = g[c] * (x[i] - mean[c]) * rsqrtf(var[c] + 1e-5f) + be[c];
  float o = mishf(v);
  if (y)  y[i]  = o;
  if (yh) yh[i] = (half_t)o;
}

__global__ void ms2fn_cvt_f16(const float* __restrict__ in, half_t* __restrict__ out, unsigned n)
{
  unsigned i = blockIdx.x * blockDim.x + threadIdx.x;
  if (i < n) out[i] = (half_t)in[i];
}

// transpose (R,C) f32 -> (C,R) f16
__global__ void ms2fn_transpose_h(const float* __restrict__ in, half_t* __restrict__ out, int R, int C)
{
  int i = blockIdx.x * blockDim.x + threadIdx.x;
  if (i >= R * C) return;
  int r = i / C, c = i - r * C;
  out[(long long)c * R + r] = (half_t)in[i];
}

// pack conv weight OIHW -> B matrix [(dy*k+dx)*C + c][o] f16
__global__ void ms2fn_pack_conv(const float* __restrict__ w, half_t* __restrict__ out,
                                int O, int C, int k)
{
  int i = blockIdx.x * blockDim.x + threadIdx.x;
  if (i >= O * C * k * k) return;
  int dx = i % k; int t = i / k; int dy = t % k; t /= k; int c = t % C; int o = t / C;
  out[(long long)((dy * k + dx) * C + c) * O + o] = (half_t)w[i];
}

// out[b,h] = sum_m dist[b,c0,m] * x[b,m,h]
__global__ void ms2fn_rowdot(const float* __restrict__ dist, const float* __restrict__ x,
                             float* __restrict__ out, int N, int Hc, int c0)
{
  int b = blockIdx.x, h = threadIdx.x;
  const float* dr = dist + (long long)b * N * N + (long long)c0 * N;
  const float* xb = x + (long long)b * N * Hc;
  float s = 0.f;
  for (int m = 0; m < N; ++m) s += dr[m] * xb[(long long)m * Hc + h];
  out[b * Hc + h] = s;
}

// depthwise: out[b,c] = sum_l x[b,l,c]*w[c,l] + bias[c]
__global__ void ms2fn_fpc(const float* __restrict__ x, const float* __restrict__ w,
                          const float* __restrict__ bias, float* __restrict__ out, int L, int Hc)
{
  int b = blockIdx.x, c = threadIdx.x;
  float s = bias[c];
  for (int l = 0; l < L; ++l) s += x[((long long)b * L + l) * Hc + c] * w[c * L + l];
  out[b * Hc + c] = s;
}

__global__ void ms2fn_avg2(const float* a, const float* b, float* o, int n)
{
  int i = blockIdx.x * blockDim.x + threadIdx.x;
  if (i < n) o[i] = 0.5f * (a[i] + b[i]);
}

__global__ void ms2fn_meanpool(const float* __restrict__ x, float* __restrict__ o, int L, int Hc)
{
  int b = blockIdx.x, h = threadIdx.x;
  float s = 0.f;
  for (int l = 0; l < L; ++l) s += x[((long long)b * L + l) * Hc + h];
  o[b * Hc + h] = s / (float)L;
}

// sim(x) row 0 applied to x:  feat[b,h] = sum_m w0[b,m]*x[b,m,h]
__global__ void ms2fn_sim_feat(const float* __restrict__ x, float* __restrict__ feat, int N, int Hc)
{
  __shared__ float sq[128], msk[128], w0[128], red[128];
  int b = blockIdx.x, t = threadIdx.x;   // blockDim = 128
  const float* xb = x + (long long)b * N * Hc;
  if (t < N) {
    float s = 0.f, ss = 0.f;
    for (int h = 0; h < Hc; ++h) { float v = xb[(long long)t * Hc + h]; s += v; ss += v * v; }
    sq[t] = ss; msk[t] = (s != 0.f) ? 1.f : 0.f;
  }
  __syncthreads();
  float e = 0.f;
  if (t < N) {
    float dot = 0.f;
    for (int h = 0; h < Hc; ++h) dot += xb[h] * xb[(long long)t * Hc + h];
    e = expf(-(sq[0] + sq[t] - 2.f * dot) / (float)Hc);
  }
  red[t] = (t < N) ? e : 0.f;
  __syncthreads();
  for (int o = 64; o > 0; o >>= 1) { if (t < o) red[t] += red[t + o]; __syncthreads(); }
  float rs = red[0];
  __syncthreads();
  if (t < N) w0[t] = (e / rs + (t == 0 ? 1.f : 0.f)) * msk[0] * msk[t];
  __syncthreads();
  float s = 0.f;
  for (int m = 0; m < N; ++m) s += w0[m] * xb[(long long)m * Hc + t];
  feat[b * Hc + t] = s;
}

__global__ void ms2fn_fuse(const float* __restrict__ feats, const float* __restrict__ fw,
                           const float* __restrict__ fb, float* __restrict__ out, unsigned BH)
{
  unsigned i = blockIdx.x * blockDim.x + threadIdx.x;
  if (i >= BH) return;
  float s = fb[0];
  #pragma unroll
  for (int k = 0; k < 6; ++k) s += fw[k] * feats[(size_t)k * BH + i];
  out[i] = s;
}

// ==================================================================
extern "C" void kernel_launch(void* const* d_in, const int* in_sizes, int n_in,
                              void* d_out, int out_size, void* d_ws, size_t ws_size,
                              hipStream_t stream)
{
  (void)in_sizes; (void)out_size; (void)ws_size;
  if (n_in < 52) return;

  const int Bn = 256, Hc = 128, Sd = 103;
  const int L1 = 9, L2 = 13, N1 = 64, N2 = 128, NCLS = 9;
  const int Pmax = Bn * L2 * L2;  // 43264

  const float* x_p1 = (const float*)d_in[0];
  const float* x_p2 = (const float*)d_in[1];
  const float* x_s1 = (const float*)d_in[2];
  const float* x_s2 = (const float*)d_in[3];
  const float* dist_p1 = (const float*)d_in[4];
  const float* dist_p2 = (const float*)d_in[5];
  const float* dist_s1 = (const float*)d_in[6];
  const float* dist_s2 = (const float*)d_in[7];

  int pi = 8;
  const float* spec_w1 = (const float*)d_in[pi++]; const float* spec_b1 = (const float*)d_in[pi++];
  const float* spec_g1 = (const float*)d_in[pi++]; const float* spec_be1 = (const float*)d_in[pi++];
  const float* spec_w2 = (const float*)d_in[pi++]; const float* spec_b2 = (const float*)d_in[pi++];
  const float* spec_g2 = (const float*)d_in[pi++]; const float* spec_be2 = (const float*)d_in[pi++];
  const float* gcn1_w = (const float*)d_in[pi++]; const float* gcn1_b = (const float*)d_in[pi++];
  const float* gbn1_g = (const float*)d_in[pi++]; const float* gbn1_be = (const float*)d_in[pi++];
  const float* gcn2_w = (const float*)d_in[pi++]; const float* gcn2_b = (const float*)d_in[pi++];
  const float* gbn2_g = (const float*)d_in[pi++]; const float* gbn2_be = (const float*)d_in[pi++];
  const float* sgcn1_w = (const float*)d_in[pi++]; const float* sgcn1_b = (const float*)d_in[pi++];
  const float* sgbn1_g = (const float*)d_in[pi++]; const float* sgbn1_be = (const float*)d_in[pi++];
  const float* sgcn2_w = (const float*)d_in[pi++]; const float* sgcn2_b = (const float*)d_in[pi++];
  const float* sgbn2_g = (const float*)d_in[pi++]; const float* sgbn2_be = (const float*)d_in[pi++];
  const float* cnn1_w = (const float*)d_in[pi++]; const float* cnn1_b = (const float*)d_in[pi++];
  const float* cbn1_g = (const float*)d_in[pi++]; const float* cbn1_be = (const float*)d_in[pi++];
  const float* cnn2_w = (const float*)d_in[pi++]; const float* cnn2_b = (const float*)d_in[pi++];
  const float* cbn2_g = (const float*)d_in[pi++]; const float* cbn2_be = (const float*)d_in[pi++];
  const float* fpc1_w = (const float*)d_in[pi++]; const float* fpc1_b = (const float*)d_in[pi++];
  const float* fpc2_w = (const float*)d_in[pi++]; const float* fpc2_b = (const float*)d_in[pi++];
  const float* fpb_g = (const float*)d_in[pi++]; const float* fpb_be = (const float*)d_in[pi++];
  const float* fus_w = (const float*)d_in[pi++]; const float* fus_b = (const float*)d_in[pi++];
  const float* bnf_g = (const float*)d_in[pi++]; const float* bnf_be = (const float*)d_in[pi++];
  const float* out_w = (const float*)d_in[pi++]; const float* out_b = (const float*)d_in[pi++];

  // ---- workspace bump allocator (needs ~200 MB) ----
  size_t off = 0;
  char* base = (char*)d_ws;
  auto alloc = [&](size_t bytes) -> char* {
    size_t a = (off + 255) & ~(size_t)255; off = a + bytes; return base + a;
  };
  half_t* w1t = (half_t*)alloc((size_t)Sd * 256 * 2);
  half_t* w2t = (half_t*)alloc((size_t)256 * 128 * 2);
  half_t* gWt = (half_t*)alloc((size_t)4 * 3 * 128 * 128 * 2);   // [set][layer]
  half_t* cw1 = (half_t*)alloc((size_t)3 * (9 * 128) * 128 * 2);
  half_t* cw2 = (half_t*)alloc((size_t)3 * (49 * 128) * 128 * 2);
  half_t* owt = (half_t*)alloc((size_t)128 * NCLS * 2);
  float* meanb = (float*)alloc(256 * 4);
  float* varb  = (float*)alloc(256 * 4);
  float* feats = (float*)alloc((size_t)6 * Bn * Hc * 4);
  float* fused = (float*)alloc((size_t)Bn * Hc * 4);
  half_t* fusedh = (half_t*)alloc((size_t)Bn * Hc * 2);
  float* xg = (float*)alloc((size_t)Bn * Hc * 4);
  float* xf = (float*)alloc((size_t)Bn * Hc * 4);
  half_t* xin_h = (half_t*)alloc((size_t)Pmax * Sd * 2);
  char*  poolA  = alloc((size_t)Pmax * 1536);          // Y1(f32+f16)  /  conv ping-pong
  float* y2f = (float*)alloc((size_t)Pmax * 128 * 4);
  half_t* y2h = (half_t*)alloc((size_t)Pmax * 128 * 2);
  float* ta_f = (float*)alloc((size_t)Pmax * 128 * 4);
  half_t* ta_h = (half_t*)alloc((size_t)Pmax * 128 * 2);
  float* tb_f = (float*)alloc((size_t)Pmax * 128 * 4);
  half_t* tb_h = (half_t*)alloc((size_t)Pmax * 128 * 2);
  half_t* dist_h = (half_t*)alloc((size_t)Bn * 169 * 169 * 2);
  // poolA views
  float*  y1f = (float*)poolA;
  half_t* y1h = (half_t*)(poolA + (size_t)Pmax * 256 * 4);
  float*  ca_f = (float*)poolA;
  float*  cb_f = (float*)(poolA + (size_t)Pmax * 128 * 4);
  half_t* ca_h = (half_t*)(poolA + (size_t)Pmax * 128 * 8);
  half_t* cb_h = (half_t*)(poolA + (size_t)Pmax * 128 * 8 + (size_t)Pmax * 128 * 2);

  auto cdiv = [](long long a, long long b) -> unsigned { return (unsigned)((a + b - 1) / b); };
  auto cvt = [&](const float* in, half_t* out, unsigned n) {
    ms2fn_cvt_f16<<<cdiv(n, 256), 256, 0, stream>>>(in, out, n);
  };
  auto gemm = [&](const half_t* A, const half_t* Bm, const float* bias, float* C, half_t* Ch,
                  int M, int N, int K, int lda, int ldb, int ldc,
                  long long sA, long long sB, long long sC, int batch,
                  int cm, int cL, int cK, int cP) {
    dim3 g((N + 63) / 64, (M + 63) / 64, batch);
    ms2fn_gemm_wmma<<<g, 256, 0, stream>>>(A, Bm, bias, C, Ch, M, N, K, lda, ldb, ldc,
                                           sA, sB, sC, cm, cL, cK, cP);
  };
  auto stats = [&](const float* x, unsigned total, int dsh, int mod) {
    ms2fn_stats<<<mod, 256, 0, stream>>>(x, meanb, varb, total, dsh, mod);
  };
  auto bnm = [&](const float* x, float* y, half_t* yh, const float* g, const float* be,
                 unsigned total, int dsh, int mod) {
    unsigned per = total / (unsigned)mod;
    dim3 gr(cdiv(per, 256), mod);
    ms2fn_bn_mish<<<gr, 256, 0, stream>>>(x, y, yh, meanb, varb, g, be, per, dsh, mod);
  };

  // ---- weight prep (f16 transposes / conv repack) ----
  ms2fn_transpose_h<<<cdiv(256 * Sd, 256), 256, 0, stream>>>(spec_w1, w1t, 256, Sd);
  ms2fn_transpose_h<<<cdiv(128 * 256, 256), 256, 0, stream>>>(spec_w2, w2t, 128, 256);
  const float* gsrc[4] = { gcn1_w, gcn2_w, sgcn1_w, sgcn2_w };
  for (int s = 0; s < 4; ++s)
    for (int i = 0; i < 3; ++i)
      ms2fn_transpose_h<<<64, 256, 0, stream>>>(gsrc[s] + (size_t)i * 16384,
                                                gWt + (size_t)(s * 3 + i) * 16384, 128, 128);
  for (int i = 0; i < 3; ++i) {
    ms2fn_pack_conv<<<cdiv(128 * 128 * 9, 256), 256, 0, stream>>>(
        cnn1_w + (size_t)i * 128 * 128 * 9, cw1 + (size_t)i * (9 * 128) * 128, 128, 128, 3);
    ms2fn_pack_conv<<<cdiv(128 * 128 * 49, 256), 256, 0, stream>>>(
        cnn2_w + (size_t)i * 128 * 128 * 49, cw2 + (size_t)i * (49 * 128) * 128, 128, 128, 7);
  }
  ms2fn_transpose_h<<<cdiv(NCLS * 128, 256), 256, 0, stream>>>(out_w, owt, NCLS, 128);

  // ---- shared branch pipeline: spectral + gcn_stack (result in tb_f / tb_h) ----
  auto run_branch = [&](const float* Xin, const float* distf, int Nn,
                        const half_t* gW, const float* gB, const float* gG, const float* gBe) {
    int P = Bn * Nn;
    cvt(Xin, xin_h, (unsigned)P * Sd);
    cvt(distf, dist_h, (unsigned)Bn * Nn * Nn);
    // spectral layer 1: (P,103)x(103,256)
    gemm(xin_h, w1t, spec_b1, y1f, nullptr, P, 256, Sd, Sd, 256, 256, 0, 0, 0, 1, 0, 0, 0, 0);
    stats(y1f, (unsigned)P * 256, 0, 256);
    bnm(y1f, y1f, y1h, spec_g1, spec_be1, (unsigned)P * 256, 0, 256);
    // spectral layer 2: (P,256)x(256,128)
    gemm(y1h, w2t, spec_b2, y2f, nullptr, P, 128, 256, 256, 128, 128, 0, 0, 0, 1, 0, 0, 0, 0);
    stats(y2f, (unsigned)P * 128, 0, 128);
    bnm(y2f, y2f, y2h, spec_g2, spec_be2, (unsigned)P * 128, 0, 128);
    // gcn stack
    const half_t* Xc = y2h;
    for (int i = 0; i < 3; ++i) {
      gemm(Xc, gW + (size_t)i * 16384, gB + i * 128, ta_f, ta_h,
           P, 128, 128, 128, 128, 128, 0, 0, 0, 1, 0, 0, 0, 0);
      gemm(dist_h, ta_h, nullptr, tb_f, nullptr, Nn, 128, Nn, Nn, 128, 128,
           (long long)Nn * Nn, (long long)Nn * 128, (long long)Nn * 128, Bn, 0, 0, 0, 0);
      stats(tb_f, (unsigned)P * 128, 7, Nn);
      bnm(tb_f, tb_f, tb_h, gG + i * Nn, gBe + i * Nn, (unsigned)P * 128, 7, Nn);
      Xc = tb_h;
    }
  };

  // ---- patch branch ----
  auto run_patch = [&](int l, int ksz, const float* Xin, const float* distf,
                       const half_t* gW, const float* gB, const float* gG, const float* gBe,
                       const half_t* cw, const float* cbias, const float* cg, const float* cbe,
                       const float* fw, const float* fbias, const float* fg, const float* fbe,
                       float* featF, float* featC) {
    int Nn = l * l, P = Bn * Nn, KK = ksz * ksz * 128;
    run_branch(Xin, distf, Nn, gW, gB, gG, gBe);
    // xg = (dist @ x)[:, center, :]
    ms2fn_rowdot<<<Bn, 128, 0, stream>>>(distf, tb_f, xg, Nn, 128, (Nn - 1) / 2);
    // xf = mish(bn(depthwise(x)));  feat = (xf + xg)/2
    ms2fn_fpc<<<Bn, 128, 0, stream>>>(tb_f, fw, fbias, xf, Nn, 128);
    stats(xf, (unsigned)Bn * 128, 0, 128);
    bnm(xf, xf, nullptr, fg, fbe, (unsigned)Bn * 128, 0, 128);
    ms2fn_avg2<<<128, 256, 0, stream>>>(xf, xg, featF, Bn * 128);
    // CNN path (implicit-GEMM convs from spectral output y2h, channels-last)
    const half_t* cin = y2h; float* lastf = nullptr;
    for (int i = 0; i < 3; ++i) {
      float* of = (i & 1) ? cb_f : ca_f;  half_t* oh = (i & 1) ? cb_h : ca_h;
      gemm(cin, cw + (size_t)i * KK * 128, cbias + i * 128, of, nullptr,
           P, 128, KK, 0, 128, 128, 0, 0, 0, 1, 1, l, ksz, ksz / 2);
      stats(of, (unsigned)P * 128, 0, 128);
      bnm(of, of, oh, cg + i * 128, cbe + i * 128, (unsigned)P * 128, 0, 128);
      cin = oh; lastf = of;
    }
    ms2fn_meanpool<<<Bn, 128, 0, stream>>>(lastf, featC, Nn, 128);
  };

  // ---- sp branch ----
  auto run_sp = [&](const float* Xin, const float* distf, int Nn,
                    const half_t* gW, const float* gB, const float* gG, const float* gBe,
                    float* feat) {
    run_branch(Xin, distf, Nn, gW, gB, gG, gBe);
    ms2fn_sim_feat<<<Bn, 128, 0, stream>>>(tb_f, feat, Nn, 128);
  };

  const unsigned BH = (unsigned)Bn * Hc;
  run_patch(L1, 3, x_p1, dist_p1, gWt + 0 * 3 * 16384, gcn1_b, gbn1_g, gbn1_be,
            cw1, cnn1_b, cbn1_g, cbn1_be, fpc1_w, fpc1_b, fpb_g + 0, fpb_be + 0,
            feats + 0 * BH, feats + 1 * BH);
  run_patch(L2, 7, x_p2, dist_p2, gWt + 1 * 3 * 16384, gcn2_b, gbn2_g, gbn2_be,
            cw2, cnn2_b, cbn2_g, cbn2_be, fpc2_w, fpc2_b, fpb_g + 128, fpb_be + 128,
            feats + 2 * BH, feats + 3 * BH);
  run_sp(x_s1, dist_s1, N1, gWt + 2 * 3 * 16384, sgcn1_b, sgbn1_g, sgbn1_be, feats + 4 * BH);
  run_sp(x_s2, dist_s2, N2, gWt + 3 * 3 * 16384, sgcn2_b, sgbn2_g, sgbn2_be, feats + 5 * BH);

  // ---- fusion + final classifier ----
  ms2fn_fuse<<<cdiv(BH, 256), 256, 0, stream>>>(feats, fus_w, fus_b, fused, BH);
  stats(fused, BH, 0, 1);
  bnm(fused, fused, fusedh, bnf_g, bnf_be, BH, 0, 1);
  gemm(fusedh, owt, out_b, (float*)d_out, nullptr, Bn, NCLS, 128, 128, NCLS, NCLS,
       0, 0, 0, 1, 0, 0, 0, 0);
}
